// BloomAttention_28707561406905
// MI455X (gfx1250) — matrix operand order repaired and checked by reading, not compiled, not run on hardware
//
#include <hip/hip_runtime.h>
#include <hip/hip_bf16.h>
#include <math.h>

// Problem constants (from reference)
constexpr int B_ = 2, S_ = 2048, D_ = 2048, H_ = 16, DH_ = 128;
constexpr int BS_ = B_ * S_;   // 4096 rows for all big GEMMs

typedef __attribute__((ext_vector_type(16))) _Float16 v16h;
typedef __attribute__((ext_vector_type(8)))  float    v8f;
typedef __attribute__((ext_vector_type(4)))  unsigned int u32x4; // trivial 16B vector
typedef __attribute__((ext_vector_type(4)))  int v4i;            // builtin's pointee type

union F16F { v16h v; u32x4 q[2]; };

// ---------------------------------------------------------------------------
// Async global -> LDS copy path (CDNA5 GLOBAL_LOAD_ASYNC_TO_LDS_B128, ASYNCcnt)
// Builtin signature (from hipcc diagnostic): (v4i addrspace(1)*, v4i addrspace(3)*, Ii, Ii)
// Falls back to a synchronous VGPR round-trip if the builtin is unavailable.
// ---------------------------------------------------------------------------
#define ASYNC_LDS_AVAILABLE 0
#if defined(__has_builtin)
#  if __has_builtin(__builtin_amdgcn_global_load_async_to_lds_b128)
#    undef ASYNC_LDS_AVAILABLE
#    define ASYNC_LDS_AVAILABLE 1
#  endif
#endif

__device__ __forceinline__ void cp16(const _Float16* g, _Float16* l) {
#if ASYNC_LDS_AVAILABLE
    __builtin_amdgcn_global_load_async_to_lds_b128(
        (__attribute__((address_space(1))) v4i*)(_Float16*)g,
        (__attribute__((address_space(3))) v4i*)l, 0, 0);
#else
    *(u32x4*)l = *(const u32x4*)g;
#endif
}

__device__ __forceinline__ void wait_async4() {
#if ASYNC_LDS_AVAILABLE
#  if __has_builtin(__builtin_amdgcn_s_wait_asynccnt)
    __builtin_amdgcn_s_wait_asynccnt(4);
#  else
    asm volatile("s_wait_asynccnt 4" ::: "memory");
#  endif
#endif
}
__device__ __forceinline__ void wait_async0() {
#if ASYNC_LDS_AVAILABLE
#  if __has_builtin(__builtin_amdgcn_s_wait_asynccnt)
    __builtin_amdgcn_s_wait_asynccnt(0);
#  else
    asm volatile("s_wait_asynccnt 0" ::: "memory");
#  endif
#endif
}

// ---------------------------------------------------------------------------
// f32 -> f16 convert (hidden states)
// ---------------------------------------------------------------------------
__global__ void k_cvt_f16(const float* __restrict__ in, _Float16* __restrict__ out, int n) {
    int i = blockIdx.x * blockDim.x + threadIdx.x;
    if (i < n) out[i] = (_Float16)in[i];
}

// f32 [K,N] -> f16 [N,K] (weights, transposed so B-fragments are contiguous)
__global__ void k_cvt_t_f16(const float* __restrict__ in, _Float16* __restrict__ out, int K, int N) {
    int i = blockIdx.x * blockDim.x + threadIdx.x;
    if (i < K * N) {
        int k = i / N, n = i % N;
        out[(size_t)n * K + k] = (_Float16)in[i];
    }
}

// ---------------------------------------------------------------------------
// Tiled WMMA GEMM:  C[M,N] = A[M,K](f16,row-major) * Bt[N,K](f16)^T + bias[N]
// MODE 0: f16 out, scatter to [B,H,S,DH]  (Q with scale, K with scale=1)
// MODE 1: f16 out, scatter to [B,H,DH,S]  (V transposed)
// MODE 2: f32 out, row-major [M,N]        (final projection -> d_out)
// Block: 256 thr (8 waves), tile 128x128, Kc=32. Wave = 64x32 (4x2 MMA tiles).
// Double-buffered LDS, async global->LDS fill overlapping WMMA compute.
// ---------------------------------------------------------------------------
template <int MODE>
__global__ __launch_bounds__(256)
void gemm_wmma_f16(const _Float16* __restrict__ A,
                   const _Float16* __restrict__ Bt,
                   const float* __restrict__ bias,
                   void* __restrict__ out,
                   int M, int N, int K, float scale) {
    __shared__ _Float16 lA[2][128 * 32];
    __shared__ _Float16 lB[2][128 * 32];

    const int tid  = threadIdx.x;
    const int lane = tid & 31;
    const int w    = tid >> 5;
    const int wm   = w >> 2;        // 0..1
    const int wn   = w & 3;         // 0..3
    const int m0   = blockIdx.x * 128;
    const int n0   = blockIdx.y * 128;
    const int lhi  = lane >> 4;     // 0/1
    const int llo  = lane & 15;

    v8f acc[4][2] = {};

    const int ldr   = tid >> 1;          // 0..127 cooperative-load row
    const int lds16 = (tid & 1) * 16;    // 0 or 16 halves

    const _Float16* gA = A  + (size_t)(m0 + ldr) * K + lds16;
    const _Float16* gB = Bt + (size_t)(n0 + ldr) * K + lds16;

    auto issue = [&](int kc, int buf) {
        cp16(gA + kc,     lA[buf] + ldr * 32 + lds16);
        cp16(gA + kc + 8, lA[buf] + ldr * 32 + lds16 + 8);
        cp16(gB + kc,     lB[buf] + ldr * 32 + lds16);
        cp16(gB + kc + 8, lB[buf] + ldr * 32 + lds16 + 8);
    };

    const int nk = K / 32;
    issue(0, 0);
    for (int i = 0; i < nk; i++) {
        __syncthreads();                 // prior compute's LDS reads done
        if (i + 1 < nk) { issue((i + 1) * 32, (i + 1) & 1); wait_async4(); }
        else            { wait_async0(); }
        __syncthreads();                 // tile i landed for all waves

        const _Float16* cA = lA[i & 1];
        const _Float16* cB = lB[i & 1];

        F16F af[4];
        for (int mt = 0; mt < 4; mt++) {
            int m = wm * 64 + mt * 16 + llo;
            const u32x4* p = (const u32x4*)(cA + m * 32 + lhi * 16);
            af[mt].q[0] = p[0];       // K = lhi*16 .. +7
            af[mt].q[1] = p[1];       // K = lhi*16 + 8 .. +15
        }
        F16F bf[2];
        for (int nt = 0; nt < 2; nt++) {
            int n = wn * 32 + nt * 16 + llo;
            const u32x4* p = (const u32x4*)(cB + n * 32 + lhi * 16);
            bf[nt].q[0] = p[0];       // K = lhi*16 .. +15 contiguous
            bf[nt].q[1] = p[1];
        }
        for (int mt = 0; mt < 4; mt++)
            for (int nt = 0; nt < 2; nt++)
                acc[mt][nt] = __builtin_amdgcn_wmma_f32_16x16x32_f16(
                    false, af[mt].v, false, bf[nt].v, (short)0, acc[mt][nt], false, false);
    }

    // Epilogue: C layout lane -> (row = j + 8*lhi, col = llo)
    for (int mt = 0; mt < 4; mt++)
        for (int nt = 0; nt < 2; nt++) {
            int col = n0 + wn * 32 + nt * 16 + llo;
            float bcol = bias[col];
            for (int j = 0; j < 8; j++) {
                int row = m0 + wm * 64 + mt * 16 + j + 8 * lhi;
                float v = (acc[mt][nt][j] + bcol) * scale;
                if (MODE == 2) {
                    ((float*)out)[(size_t)row * N + col] = v;
                } else {
                    int b = row / S_, s = row % S_;
                    int h = col / DH_, dh = col % DH_;
                    _Float16 hv = (_Float16)v;
                    if (MODE == 0)
                        ((_Float16*)out)[(((size_t)(b * H_ + h)) * S_ + s) * DH_ + dh] = hv;
                    else
                        ((_Float16*)out)[(((size_t)(b * H_ + h)) * DH_ + dh) * S_ + s] = hv;
                }
            }
        }
}

// ---------------------------------------------------------------------------
// Flash attention with ALiBi + mask.
// Block = 256 thr (8 waves) covers 128 query rows of one (b,h); each wave owns
// a 16-row tile. Key loop in chunks of 32; K/V chunks double-buffered in LDS
// and filled via async global->LDS, overlapping the WMMA + softmax compute.
// Q pre-scaled by 1/sqrt(DH). Output: context f16 [B, S, H*DH].
// ---------------------------------------------------------------------------
__global__ __launch_bounds__(256)
void flash_attn(const _Float16* __restrict__ Q,   // [B,H,S,DH] (scaled)
                const _Float16* __restrict__ Kk,  // [B,H,S,DH]
                const _Float16* __restrict__ Vt,  // [B,H,DH,S]
                const unsigned char* __restrict__ mask, // [B,S,S] bool
                _Float16* __restrict__ ctx) {     // [B,S,H*DH]
    __shared__ _Float16 lK[2][32 * 128];  // [key][dh]
    __shared__ _Float16 lV[2][128 * 32];  // [dh][key]
    __shared__ _Float16 lP[8][16 * 32];   // per-wave P tile

    const int tid  = threadIdx.x;
    const int lane = tid & 31;
    const int w    = tid >> 5;
    const int lhi  = lane >> 4, llo = lane & 15;

    const int nqb = S_ / 128;
    const int bh  = blockIdx.x / nqb;
    const int qb  = blockIdx.x % nqb;
    const int b   = bh / H_, h = bh % H_;
    const int sq0 = qb * 128 + w * 16;

    // BLOOM ALiBi slope for head h
    float slope;
    {
        int closest = 1;
        while ((closest << 1) <= H_) closest <<= 1;
        float base = exp2f(-exp2f(-(log2f((float)closest) - 3.0f)));
        if (h < closest) slope = powf(base, (float)(h + 1));
        else {
            float eb = exp2f(-exp2f(-(log2f(2.0f * (float)closest) - 3.0f)));
            slope = powf(eb, (float)(1 + 2 * (h - closest)));
        }
    }

    // Load Q fragments once (A-layout over DH=128 -> 4 frags of K=32)
    F16F qf[4];
    {
        const _Float16* qp = Q + (((size_t)(b * H_ + h)) * S_ + (sq0 + llo)) * DH_;
        for (int f = 0; f < 4; f++) {
            const u32x4* p = (const u32x4*)(qp + f * 32 + lhi * 16);
            qf[f].q[0] = p[0];
            qf[f].q[1] = p[1];
        }
    }

    v8f O[8] = {};
    float mrow[8], lrow[8];
    for (int j = 0; j < 8; j++) { mrow[j] = -INFINITY; lrow[j] = 0.0f; }

    const _Float16* Kbase = Kk + ((size_t)(b * H_ + h)) * S_ * DH_;
    const _Float16* Vbase = Vt + ((size_t)(b * H_ + h)) * DH_ * S_;
    const unsigned char* mb = mask + (size_t)b * S_ * S_;

    const int vr = tid >> 1, vseg = (tid & 1) * 16;   // V chunk: 128 rows x 2 halves
    const int kr = tid >> 3, kseg = (tid & 7) * 16;   // K chunk: 32 rows x 8 segs

    auto issue = [&](int kb, int buf) {
        cp16(Kbase + (size_t)(kb + kr) * DH_ + kseg,     lK[buf] + kr * 128 + kseg);
        cp16(Kbase + (size_t)(kb + kr) * DH_ + kseg + 8, lK[buf] + kr * 128 + kseg + 8);
        cp16(Vbase + (size_t)vr * S_ + kb + vseg,        lV[buf] + vr * 32 + vseg);
        cp16(Vbase + (size_t)vr * S_ + kb + vseg + 8,    lV[buf] + vr * 32 + vseg + 8);
    };

    const int nch = S_ / 32;
    issue(0, 0);
    for (int i = 0; i < nch; i++) {
        const int kb = i * 32;
        __syncthreads();                 // prior compute's LDS reads done
        if (i + 1 < nch) { issue((i + 1) * 32, (i + 1) & 1); wait_async4(); }
        else             { wait_async0(); }
        __syncthreads();                 // chunk i landed for all waves

        const _Float16* cK = lK[i & 1];
        const _Float16* cV = lV[i & 1];

        // Scores: two 16-col tiles, each = 4 WMMA over DH
        float sc[2][8];
        for (int nt = 0; nt < 2; nt++) {
            v8f a = {};
            for (int f = 0; f < 4; f++) {
                F16F kf;
                const u32x4* p = (const u32x4*)(cK + (nt * 16 + llo) * 128 + f * 32 + lhi * 16);
                kf.q[0] = p[0]; kf.q[1] = p[1];
                a = __builtin_amdgcn_wmma_f32_16x16x32_f16(
                        false, qf[f].v, false, kf.v, (short)0, a, false, false);
            }
            int sk = kb + nt * 16 + llo;
            float ab = slope * (float)(sk - (S_ - 1));
            const unsigned char* mp = mb + sk;
            for (int j = 0; j < 8; j++) {
                int sq = sq0 + j + 8 * lhi;
                bool ok = mp[(size_t)sq * S_] != 0;
                sc[nt][j] = ok ? (a[j] + ab) : -1e9f;
            }
        }

        // Online softmax (row = j + 8*lhi lives across 16 lanes: xor 1,2,4,8)
        for (int j = 0; j < 8; j++) {
            float mx = fmaxf(sc[0][j], sc[1][j]);
            mx = fmaxf(mx, __shfl_xor(mx, 1));
            mx = fmaxf(mx, __shfl_xor(mx, 2));
            mx = fmaxf(mx, __shfl_xor(mx, 4));
            mx = fmaxf(mx, __shfl_xor(mx, 8));
            float mnew  = fmaxf(mrow[j], mx);
            float alpha = __expf(mrow[j] - mnew);
            float p0 = __expf(sc[0][j] - mnew);
            float p1 = __expf(sc[1][j] - mnew);
            float rs = p0 + p1;
            rs += __shfl_xor(rs, 1);
            rs += __shfl_xor(rs, 2);
            rs += __shfl_xor(rs, 4);
            rs += __shfl_xor(rs, 8);
            lrow[j] = lrow[j] * alpha + rs;
            mrow[j] = mnew;
            for (int t = 0; t < 8; t++) O[t][j] *= alpha;
            sc[0][j] = p0; sc[1][j] = p1;
        }

        // Re-stripe P: C-layout -> A-layout via wave-private LDS
        {
            _Float16* pw = lP[w];
            for (int j = 0; j < 8; j++) {
                int r = j + 8 * lhi;
                pw[r * 32 + llo]      = (_Float16)sc[0][j];
                pw[r * 32 + 16 + llo] = (_Float16)sc[1][j];
            }
        }
        F16F pf;
        {
            const u32x4* p = (const u32x4*)(lP[w] + llo * 32 + lhi * 16);
            pf.q[0] = p[0];
            pf.q[1] = p[1];
        }

        // O += P x V : 8 WMMA over DH tiles
        for (int dt = 0; dt < 8; dt++) {
            F16F vf;
            const u32x4* p = (const u32x4*)(cV + (dt * 16 + llo) * 32 + lhi * 16);
            vf.q[0] = p[0]; vf.q[1] = p[1];
            O[dt] = __builtin_amdgcn_wmma_f32_16x16x32_f16(
                        false, pf.v, false, vf.v, (short)0, O[dt], false, false);
        }
    }

    // Finalize: divide by row sums, write context [B,S,H*DH] f16
    for (int j = 0; j < 8; j++) {
        float inv = 1.0f / lrow[j];
        int sq = sq0 + j + 8 * lhi;
        _Float16* cp = ctx + ((size_t)b * S_ + sq) * (H_ * DH_) + h * DH_;
        for (int dt = 0; dt < 8; dt++)
            cp[dt * 16 + llo] = (_Float16)(O[dt][j] * inv);
    }
}

// ---------------------------------------------------------------------------
extern "C" void kernel_launch(void* const* d_in, const int* in_sizes, int n_in,
                              void* d_out, int out_size, void* d_ws, size_t ws_size,
                              hipStream_t stream) {
    const float*         hidden = (const float*)d_in[0];
    const unsigned char* mask   = (const unsigned char*)d_in[1]; // bool -> 1 byte
    const float* wq = (const float*)d_in[2];
    const float* bq = (const float*)d_in[3];
    const float* wk = (const float*)d_in[4];
    const float* bk = (const float*)d_in[5];
    const float* wv = (const float*)d_in[6];
    const float* bv = (const float*)d_in[7];
    const float* wo = (const float*)d_in[8];
    const float* bo = (const float*)d_in[9];
    float* out = (float*)d_out;

    // Workspace carve-up (f16 elements): ~117 MB total
    _Float16* ws = (_Float16*)d_ws;
    size_t o = 0;
    _Float16* h16 = ws + o; o += (size_t)B_ * S_ * D_;
    _Float16* wqT = ws + o; o += (size_t)D_ * D_;
    _Float16* wkT = ws + o; o += (size_t)D_ * D_;
    _Float16* wvT = ws + o; o += (size_t)D_ * D_;
    _Float16* woT = ws + o; o += (size_t)D_ * D_;
    _Float16* Qh  = ws + o; o += (size_t)B_ * H_ * S_ * DH_;
    _Float16* Kh  = ws + o; o += (size_t)B_ * H_ * S_ * DH_;
    _Float16* VTh = ws + o; o += (size_t)B_ * H_ * S_ * DH_;
    _Float16* CTX = ws + o; o += (size_t)B_ * S_ * D_;

    // 1) Convert activations + transpose-convert weights to f16
    int nh = B_ * S_ * D_;
    k_cvt_f16<<<(nh + 255) / 256, 256, 0, stream>>>(hidden, h16, nh);
    int nw = D_ * D_;
    int gw = (nw + 255) / 256;
    k_cvt_t_f16<<<gw, 256, 0, stream>>>(wq, wqT, D_, D_);
    k_cvt_t_f16<<<gw, 256, 0, stream>>>(wk, wkT, D_, D_);
    k_cvt_t_f16<<<gw, 256, 0, stream>>>(wv, wvT, D_, D_);
    k_cvt_t_f16<<<gw, 256, 0, stream>>>(wo, woT, D_, D_);

    // 2) QKV projections (WMMA GEMMs); fold 1/sqrt(DH) into Q
    dim3 g(BS_ / 128, D_ / 128);
    float invnorm = 1.0f / sqrtf((float)DH_);
    gemm_wmma_f16<0><<<g, 256, 0, stream>>>(h16, wqT, bq, (void*)Qh,  BS_, D_, D_, invnorm);
    gemm_wmma_f16<0><<<g, 256, 0, stream>>>(h16, wkT, bk, (void*)Kh,  BS_, D_, D_, 1.0f);
    gemm_wmma_f16<1><<<g, 256, 0, stream>>>(h16, wvT, bv, (void*)VTh, BS_, D_, D_, 1.0f);

    // 3) Flash attention with ALiBi + mask -> context f16
    flash_attn<<<B_ * H_ * (S_ / 128), 256, 0, stream>>>(Qh, Kh, VTh, mask, CTX);

    // 4) Output projection -> f32 d_out
    gemm_wmma_f16<2><<<g, 256, 0, stream>>>(CTX, woT, bo, (void*)out, BS_, D_, D_, 1.0f);
}
